// ResidualLSTM_38749194944731
// MI455X (gfx1250) — matrix operand order repaired and checked by reading, not compiled, not run on hardware
//
#include <hip/hip_runtime.h>
#include <cstdint>
#include <cstddef>

// ---------------------------------------------------------------------------
// Residual 2-layer LSTM for MI455X (gfx1250, wave32, WMMA).
// T=512, B=1024, IN=32, H0=128, H1=64.
//
// Strategy: batch-parallel recurrence. Weights converted once to bf16 and
// pre-packed into V_WMMA_F32_16X16X32_BF16 B-fragment order in d_ws, then
// cached in LDS (~280KB) by each workgroup. 16 WGs x 4 waves; each wave owns
// a 16-row batch tile and iterates all 512 time steps with h/c state held in
// registers in WMMA A-/C-fragment layouts. A small per-wave LDS staging
// buffer converts D-layout (f32) hidden outputs back into A-layout bf16
// fragments for the next step's recurrent matmul.
//
// Round 2 change: use native float<->__bf16 conversions (hardware
// V_CVT_*BF16* path) instead of manual RNE bit-twiddling, shortening the
// serial per-step dependency chain and cutting VALU count.
// ---------------------------------------------------------------------------

#define T_STEPS 512
#define BATCH   1024
#define IN_DIM  32
#define H0_DIM  128
#define H1_DIM  64

typedef __bf16 bf16;
typedef __attribute__((ext_vector_type(16))) __bf16       v16bf;
typedef __attribute__((ext_vector_type(8)))  float        v8f;
typedef __attribute__((ext_vector_type(4)))  unsigned int u32x4;

// Packed-weight layout (bf16 elements) inside d_ws:
//   index = (((nt*KB) + kb)*32 + lane)*16 + v
//   K = kb*32 + (v&8)*2 + ((lane>>4)&1)*8 + (v&7)   (B-fragment K mapping, ISA 7.12.2)
//   N = nt*16 + (lane&15)
#define OFF_WIH0   0        // KB=1, NT=32 -> 16384
#define OFF_WHH0   16384    // KB=4, NT=32 -> 65536
#define OFF_WS0    81920    // KB=1, NT=8  -> 4096
#define OFF_WIH1   86016    // KB=4, NT=16 -> 32768
#define OFF_WHH1   118784   // KB=2, NT=16 -> 16384
#define OFF_WS1    135168   // KB=4, NT=4  -> 8192
#define W_ELEMS    143360   // total bf16 elements (286720 bytes)
#define W_BYTES    (W_ELEMS * 2)
#define B_FLOATS   960      // bg0[512] bs0[128] bg1[256] bs1[64]
#define B_BYTES    (B_FLOATS * 4)
#define STAGE_ELEMS_PER_WAVE (16*H0_DIM + 16*H1_DIM)   // 3072 bf16
#define SMEM_BYTES (W_BYTES + B_BYTES + 4*STAGE_ELEMS_PER_WAVE*2)  // 315136

// ---------------- bf16 <-> f32 (native hardware conversion) ----------------
static __device__ __forceinline__ bf16 f2bf(float f) {
    return (bf16)f;      // fptrunc -> v_cvt_(pk_)bf16_f32
}
static __device__ __forceinline__ float bf2f(bf16 b) {
    return (float)b;     // fpext -> shift/cvt
}

static __device__ __forceinline__ float sigm(float x) {
    return 1.0f / (1.0f + __expf(-x));
}

static __device__ __forceinline__ v8f splat8(float v) {
    v8f r;
#pragma unroll
    for (int i = 0; i < 8; ++i) r[i] = v;
    return r;
}

// ---------------- WMMA wrapper --------------------------------------------
static __device__ __forceinline__ v8f wmma_bf16(v16bf a, v16bf b, v8f c) {
    // D = A(16x32 bf16) * B(32x16 bf16) + C(16x16 f32)
    return __builtin_amdgcn_wmma_f32_16x16x32_bf16(
        /*neg_a=*/false, a, /*neg_b=*/false, b,
        /*c_mod=*/(short)0, c, /*reuse_a=*/false, /*reuse_b=*/false);
}

// ---------------- fragment loaders ----------------------------------------
// A-fragment (16x32 bf16) from row-major bf16 [16][ldK] in LDS.
// Per ISA: lane M = lane&15; lanes<16 hold K 0-7,16-23; lanes>=16 hold 8-15,24-31.
static __device__ __forceinline__ v16bf ldsA(const bf16* base, int ldK, int lane, int kb) {
    const int row = lane & 15;
    const int k0  = kb * 32 + ((lane >> 4) & 1) * 8;
    const bf16* p = base + row * ldK + k0;
    union { u32x4 q[2]; v16bf v; } u;
    u.q[0] = *(const u32x4*)(p);        // K = k0 .. k0+7
    u.q[1] = *(const u32x4*)(p + 16);   // K = k0+16 .. k0+23
    return u.v;
}

// A-fragment from row-major f32 [rows][ldK] in global memory (x and h init).
static __device__ __forceinline__ v16bf gldA(const float* base, int ldK, int lane, int kb) {
    const int row = lane & 15;
    const int k0  = kb * 32 + ((lane >> 4) & 1) * 8;
    const float* p = base + row * ldK + k0;
    v16bf a;
#pragma unroll
    for (int v = 0; v < 8; ++v) a[v] = f2bf(p[v]);
#pragma unroll
    for (int v = 0; v < 8; ++v) a[8 + v] = f2bf(p[16 + v]);
    return a;
}

// B-fragment from pre-packed LDS weights (32 contiguous bytes per lane).
static __device__ __forceinline__ v16bf ldsB(const bf16* packed, int KB, int nt, int kb, int lane) {
    const bf16* p = packed + (size_t)(((nt * KB) + kb) * 32 + lane) * 16;
    union { u32x4 q[2]; v16bf v; } u;
    u.q[0] = *(const u32x4*)(p);
    u.q[1] = *(const u32x4*)(p + 8);
    return u.v;
}

static __device__ __forceinline__ v16bf clipA(v16bf a) {
    v16bf r;
#pragma unroll
    for (int i = 0; i < 16; ++i) {
        float f = bf2f(a[i]);
        f = fminf(10.0f, fmaxf(-10.0f, f));
        r[i] = f2bf(f);
    }
    return r;
}

// C/D fragment (16x16 f32): vgpr r -> M = ((lane>>4)&1)*8 + r, N = lane&15.
static __device__ __forceinline__ v8f gldC(const float* base, int ldN, int row0, int nt, int lane) {
    const int n  = nt * 16 + (lane & 15);
    const int mb = ((lane >> 4) & 1) * 8;
    v8f c;
#pragma unroll
    for (int r = 0; r < 8; ++r) c[r] = base[(size_t)(row0 + mb + r) * ldN + n];
    return c;
}
static __device__ __forceinline__ void gstC(float* base, int ldN, int row0, int nt, int lane, v8f c) {
    const int n  = nt * 16 + (lane & 15);
    const int mb = ((lane >> 4) & 1) * 8;
#pragma unroll
    for (int r = 0; r < 8; ++r) base[(size_t)(row0 + mb + r) * ldN + n] = c[r];
}

// ---------------- prep kernels --------------------------------------------
// Transpose + bf16-convert weights into B-fragment-linear order.
// W is [Nout][Kin] row-major (PyTorch convention); B element (k,n) = W[n][k].
__global__ void pack_weight(const float* __restrict__ W, bf16* __restrict__ dst,
                            int Kin, int Nout) {
    int idx = blockIdx.x * blockDim.x + threadIdx.x;
    int KB = Kin >> 5;
    int total = KB * (Nout >> 4) * 512;
    if (idx >= total) return;
    int v    = idx & 15;
    int lane = (idx >> 4) & 31;
    int t2   = idx >> 9;
    int kb   = t2 % KB;
    int nt   = t2 / KB;
    int k = kb * 32 + (v & 8) * 2 + ((lane >> 4) & 1) * 8 + (v & 7);
    int n = nt * 16 + (lane & 15);
    dst[idx] = f2bf(W[(size_t)n * Kin + k]);
}

__global__ void pack_bias(const float* __restrict__ bih0, const float* __restrict__ bhh0,
                          const float* __restrict__ bs0,
                          const float* __restrict__ bih1, const float* __restrict__ bhh1,
                          const float* __restrict__ bs1, float* __restrict__ dst) {
    int i = blockIdx.x * blockDim.x + threadIdx.x;
    if (i < 512)      dst[i] = bih0[i] + bhh0[i];
    else if (i < 640) dst[i] = bs0[i - 512];
    else if (i < 896) dst[i] = bih1[i - 640] + bhh1[i - 640];
    else if (i < 960) dst[i] = bs1[i - 896];
}

// ---------------- main recurrence kernel ----------------------------------
__global__ __launch_bounds__(128) void
residual_lstm_kernel(const float* __restrict__ x,
                     const float* __restrict__ h0i, const float* __restrict__ c0i,
                     const float* __restrict__ h1i, const float* __restrict__ c1i,
                     const bf16*  __restrict__ wpack,
                     const float* __restrict__ bpack,
                     float* __restrict__ ys,
                     float* __restrict__ h0o, float* __restrict__ c0o,
                     float* __restrict__ h1o, float* __restrict__ c1o) {
    extern __shared__ unsigned char smem[];
    bf16*  sW     = (bf16*)smem;
    float* sBias  = (float*)(smem + W_BYTES);
    bf16*  sStage = (bf16*)(smem + W_BYTES + B_BYTES);

    const int tid  = threadIdx.x;
    const int wave = tid >> 5;
    const int lane = tid & 31;
    const int lc   = lane & 15;
    const int mb   = ((lane >> 4) & 1) * 8;

    // Cooperative LDS fill: all packed weights + biases.
    {
        const u32x4* src = (const u32x4*)wpack;
        u32x4* dst = (u32x4*)sW;
        for (int i = tid; i < W_BYTES / 16; i += 128) dst[i] = src[i];
        for (int i = tid; i < B_FLOATS; i += 128) sBias[i] = bpack[i];
    }
    __syncthreads();

    bf16* stage0 = sStage + wave * STAGE_ELEMS_PER_WAVE;     // [16][128] bf16
    bf16* stage1 = stage0 + 16 * H0_DIM;                      // [16][64]  bf16

    const bf16* Wih0p = sW + OFF_WIH0;
    const bf16* Whh0p = sW + OFF_WHH0;
    const bf16* Ws0p  = sW + OFF_WS0;
    const bf16* Wih1p = sW + OFF_WIH1;
    const bf16* Whh1p = sW + OFF_WHH1;
    const bf16* Ws1p  = sW + OFF_WS1;
    const float* bg0  = sBias;            // 512: b_ih0 + b_hh0
    const float* bsl0 = sBias + 512;      // 128: bs0
    const float* bg1  = sBias + 640;      // 256: b_ih1 + b_hh1
    const float* bsl1 = sBias + 896;      // 64 : bs1

    const int row0 = (blockIdx.x * 4 + wave) * 16;   // batch tile base row

    // Persistent state in WMMA fragment layouts.
    v16bf h0f[4], h1f[2], inp1f[4];
    v8f   c0f[8], c1f[4];

#pragma unroll
    for (int kb = 0; kb < 4; ++kb) h0f[kb] = gldA(h0i + (size_t)row0 * H0_DIM, H0_DIM, lane, kb);
#pragma unroll
    for (int kb = 0; kb < 2; ++kb) h1f[kb] = gldA(h1i + (size_t)row0 * H1_DIM, H1_DIM, lane, kb);
#pragma unroll
    for (int nt = 0; nt < 8; ++nt) c0f[nt] = gldC(c0i, H0_DIM, row0, nt, lane);
#pragma unroll
    for (int nt = 0; nt < 4; ++nt) c1f[nt] = gldC(c1i, H1_DIM, row0, nt, lane);

    for (int t = 0; t < T_STEPS; ++t) {
        const float* xb = x + ((size_t)t * BATCH + row0) * IN_DIM;
        v16bf xf = gldA(xb, IN_DIM, lane, 0);   // 16x32 input tile

        // ---------------- layer 0: gates N=512 (i|f|g|o x 128) ------------
#pragma unroll
        for (int j = 0; j < 8; ++j) {           // hidden tile j: cols j*16..j*16+15
            v8f gi = splat8(bg0[  0 + j * 16 + lc]);
            v8f gf = splat8(bg0[128 + j * 16 + lc]);
            v8f gg = splat8(bg0[256 + j * 16 + lc]);
            v8f go = splat8(bg0[384 + j * 16 + lc]);
            // x @ W_ih0.T  (K=32)
            gi = wmma_bf16(xf, ldsB(Wih0p, 1, j     , 0, lane), gi);
            gf = wmma_bf16(xf, ldsB(Wih0p, 1, j +  8, 0, lane), gf);
            gg = wmma_bf16(xf, ldsB(Wih0p, 1, j + 16, 0, lane), gg);
            go = wmma_bf16(xf, ldsB(Wih0p, 1, j + 24, 0, lane), go);
            // h0 @ W_hh0.T (K=128)
#pragma unroll
            for (int kb = 0; kb < 4; ++kb) {
                gi = wmma_bf16(h0f[kb], ldsB(Whh0p, 4, j     , kb, lane), gi);
                gf = wmma_bf16(h0f[kb], ldsB(Whh0p, 4, j +  8, kb, lane), gf);
                gg = wmma_bf16(h0f[kb], ldsB(Whh0p, 4, j + 16, kb, lane), gg);
                go = wmma_bf16(h0f[kb], ldsB(Whh0p, 4, j + 24, kb, lane), go);
            }
            // shortcut: x @ Ws0.T + bs0
            v8f rs = splat8(bsl0[j * 16 + lc]);
            rs = wmma_bf16(xf, ldsB(Ws0p, 1, j, 0, lane), rs);

            // cell + residual (elementwise on C/D layout)
            v8f c = c0f[j], hn;
#pragma unroll
            for (int r = 0; r < 8; ++r) {
                float iv = sigm(gi[r]);
                float fv = sigm(gf[r]);
                float gv = tanhf(gg[r]);
                float ov = sigm(go[r]);
                float cn = fv * c[r] + iv * gv;
                c[r]  = cn;
                hn[r] = ov * tanhf(cn) + rs[r];   // post-residual, PRE-clip
            }
            c0f[j] = c;
            // stage bf16 h0 (pre-clip) for A-fragment reload
#pragma unroll
            for (int r = 0; r < 8; ++r)
                stage0[(mb + r) * H0_DIM + j * 16 + lc] = f2bf(hn[r]);
            if (t == T_STEPS - 1) {
#pragma unroll
                for (int r = 0; r < 8; ++r)
                    h0o[(size_t)(row0 + mb + r) * H0_DIM + j * 16 + lc] = hn[r];
            }
        }

        // Reload carried h0 (raw) and layer-1 input (clipped) as A-fragments.
#pragma unroll
        for (int kb = 0; kb < 4; ++kb) {
            v16bf raw = ldsA(stage0, H0_DIM, lane, kb);
            h0f[kb]   = raw;
            inp1f[kb] = clipA(raw);
        }

        // ---------------- layer 1: gates N=256 (i|f|g|o x 64) -------------
#pragma unroll
        for (int j = 0; j < 4; ++j) {
            v8f gi = splat8(bg1[  0 + j * 16 + lc]);
            v8f gf = splat8(bg1[ 64 + j * 16 + lc]);
            v8f gg = splat8(bg1[128 + j * 16 + lc]);
            v8f go = splat8(bg1[192 + j * 16 + lc]);
            // inp1 @ W_ih1.T (K=128)
#pragma unroll
            for (int kb = 0; kb < 4; ++kb) {
                gi = wmma_bf16(inp1f[kb], ldsB(Wih1p, 4, j     , kb, lane), gi);
                gf = wmma_bf16(inp1f[kb], ldsB(Wih1p, 4, j +  4, kb, lane), gf);
                gg = wmma_bf16(inp1f[kb], ldsB(Wih1p, 4, j +  8, kb, lane), gg);
                go = wmma_bf16(inp1f[kb], ldsB(Wih1p, 4, j + 12, kb, lane), go);
            }
            // h1 @ W_hh1.T (K=64)
#pragma unroll
            for (int kb = 0; kb < 2; ++kb) {
                gi = wmma_bf16(h1f[kb], ldsB(Whh1p, 2, j     , kb, lane), gi);
                gf = wmma_bf16(h1f[kb], ldsB(Whh1p, 2, j +  4, kb, lane), gf);
                gg = wmma_bf16(h1f[kb], ldsB(Whh1p, 2, j +  8, kb, lane), gg);
                go = wmma_bf16(h1f[kb], ldsB(Whh1p, 2, j + 12, kb, lane), go);
            }
            // shortcut: inp1 @ Ws1.T + bs1 (K=128)
            v8f rs = splat8(bsl1[j * 16 + lc]);
#pragma unroll
            for (int kb = 0; kb < 4; ++kb)
                rs = wmma_bf16(inp1f[kb], ldsB(Ws1p, 4, j, kb, lane), rs);

            v8f c = c1f[j], hn;
#pragma unroll
            for (int r = 0; r < 8; ++r) {
                float iv = sigm(gi[r]);
                float fv = sigm(gf[r]);
                float gv = tanhf(gg[r]);
                float ov = sigm(go[r]);
                float cn = fv * c[r] + iv * gv;
                c[r]  = cn;
                hn[r] = ov * tanhf(cn) + rs[r];   // carried h1 is PRE-clip
            }
            c1f[j] = c;
#pragma unroll
            for (int r = 0; r < 8; ++r) {
                float yv = fminf(10.0f, fmaxf(-10.0f, hn[r]));
                ys[((size_t)t * BATCH + row0 + mb + r) * H1_DIM + j * 16 + lc] = yv;
                stage1[(mb + r) * H1_DIM + j * 16 + lc] = f2bf(hn[r]);
            }
            if (t == T_STEPS - 1) {
#pragma unroll
                for (int r = 0; r < 8; ++r)
                    h1o[(size_t)(row0 + mb + r) * H1_DIM + j * 16 + lc] = hn[r];
            }
        }
#pragma unroll
        for (int kb = 0; kb < 2; ++kb) h1f[kb] = ldsA(stage1, H1_DIM, lane, kb);
    }

    // Final cell states (exact fp32 from accumulators).
#pragma unroll
    for (int nt = 0; nt < 8; ++nt) gstC(c0o, H0_DIM, row0, nt, lane, c0f[nt]);
#pragma unroll
    for (int nt = 0; nt < 4; ++nt) gstC(c1o, H1_DIM, row0, nt, lane, c1f[nt]);
}

// ---------------------------------------------------------------------------
extern "C" void kernel_launch(void* const* d_in, const int* in_sizes, int n_in,
                              void* d_out, int out_size, void* d_ws, size_t ws_size,
                              hipStream_t stream) {
    (void)in_sizes; (void)n_in; (void)out_size; (void)ws_size;

    const float* x    = (const float*)d_in[0];
    const float* h0i  = (const float*)d_in[1];
    const float* c0i  = (const float*)d_in[2];
    const float* h1i  = (const float*)d_in[3];
    const float* c1i  = (const float*)d_in[4];
    const float* Wih0 = (const float*)d_in[5];
    const float* Whh0 = (const float*)d_in[6];
    const float* bih0 = (const float*)d_in[7];
    const float* bhh0 = (const float*)d_in[8];
    const float* Ws0  = (const float*)d_in[9];
    const float* bs0  = (const float*)d_in[10];
    const float* Wih1 = (const float*)d_in[11];
    const float* Whh1 = (const float*)d_in[12];
    const float* bih1 = (const float*)d_in[13];
    const float* bhh1 = (const float*)d_in[14];
    const float* Ws1  = (const float*)d_in[15];
    const float* bs1  = (const float*)d_in[16];

    bf16*  wp = (bf16*)d_ws;
    float* bp = (float*)((char*)d_ws + W_BYTES);

    pack_weight<<<(16384 + 255) / 256, 256, 0, stream>>>(Wih0, wp + OFF_WIH0,  32, 512);
    pack_weight<<<(65536 + 255) / 256, 256, 0, stream>>>(Whh0, wp + OFF_WHH0, 128, 512);
    pack_weight<<<( 4096 + 255) / 256, 256, 0, stream>>>(Ws0 , wp + OFF_WS0 ,  32, 128);
    pack_weight<<<(32768 + 255) / 256, 256, 0, stream>>>(Wih1, wp + OFF_WIH1, 128, 256);
    pack_weight<<<(16384 + 255) / 256, 256, 0, stream>>>(Whh1, wp + OFF_WHH1,  64, 256);
    pack_weight<<<( 8192 + 255) / 256, 256, 0, stream>>>(Ws1 , wp + OFF_WS1 , 128,  64);
    pack_bias<<<4, 256, 0, stream>>>(bih0, bhh0, bs0, bih1, bhh1, bs1, bp);

    float* out = (float*)d_out;
    float* ysp = out;                                        // [512,1024,64]
    float* h0o = ysp + (size_t)T_STEPS * BATCH * H1_DIM;     // [1024,128]
    float* c0o = h0o + (size_t)BATCH * H0_DIM;               // [1024,128]
    float* h1o = c0o + (size_t)BATCH * H0_DIM;               // [1024,64]
    float* c1o = h1o + (size_t)BATCH * H1_DIM;               // [1024,64]

    (void)hipFuncSetAttribute((const void*)residual_lstm_kernel,
                              hipFuncAttributeMaxDynamicSharedMemorySize, SMEM_BYTES);
    residual_lstm_kernel<<<dim3(16), dim3(128), SMEM_BYTES, stream>>>(
        x, h0i, c0i, h1i, c1i, wp, bp, ysp, h0o, c0o, h1o, c1o);
}
